// CrossAttnBlockpp_68504728371760
// MI455X (gfx1250) — matrix operand order repaired and checked by reading, not compiled, not run on hardware
//
#include <hip/hip_runtime.h>

typedef __bf16 bf16;
typedef __attribute__((ext_vector_type(16))) __bf16 v16bf;
typedef __attribute__((ext_vector_type(8)))  __bf16 v8bf;
typedef __attribute__((ext_vector_type(4)))  __bf16 v4bf;
typedef __attribute__((ext_vector_type(8)))  float  v8f;
typedef __attribute__((ext_vector_type(4)))  unsigned int u32x4;

#define NB 4
#define NC 128
#define NL 4096          // 64*64
#define NG 32
#define CPG 4
#define RESULT_ELEMS (NB * 2 * NC * NL)
#define ATT_SCALE 0.08838834764831845f   // 128^-0.5
#define INV_SQRT2 0.7071067811865475f

// A-fragment (16x32 bf16) K index for half-slot h (ISA 7.12.2): lane<16 holds
// K{0..7,16..23}, lane>=16 holds K{8..15,24..31}
__device__ __forceinline__ int a_k(int h, int hi) {
    return (h < 8 ? h : h + 8) + (hi ? 8 : 0);
}

// two 16B loads -> one 16-element bf16 fragment (avoids 32B-align requirement)
__device__ __forceinline__ v16bf ld2x8(const bf16* lo, const bf16* hi8) {
    v8bf a = *(const v8bf*)lo;
    v8bf b = *(const v8bf*)hi8;
    return __builtin_shufflevector(a, b, 0, 1, 2, 3, 4, 5, 6, 7,
                                   8, 9, 10, 11, 12, 13, 14, 15);
}
// 16 contiguous bf16 (16B aligned)
__device__ __forceinline__ v16bf ld16(const bf16* p) { return ld2x8(p, p + 8); }

// ---------------------------------------------------------------- weights -> bf16, transposed
// dstT[mat][d][c] = w[c][d]  (so B-fragments read contiguous rows)
__global__ void wconv_kernel(const float* w0, const float* w1, const float* w2,
                             const float* w3, const float* w4, const float* w5,
                             bf16* __restrict__ dstT) {
    const float* src;
    switch (blockIdx.y) {
        case 0: src = w0; break; case 1: src = w1; break;
        case 2: src = w2; break; case 3: src = w3; break;
        case 4: src = w4; break; default: src = w5; break;
    }
    int i = blockIdx.x * 256 + threadIdx.x;   // 0..16383
    int d = i >> 7, c = i & 127;
    dstT[(size_t)blockIdx.y * 16384 + i] = (bf16)src[c * 128 + d];
}

// ---------------------------------------------------------------- group norm -> bf16 (L,C)
// Each block owns one (b, half, group) = 4 consecutive channels over all 4096 l.
// Stats pass reads the contiguous 16384-float group; normalize pass transposes
// on the fly: one packed 8B store per l covering the group's 4 channels.
__global__ __launch_bounds__(256) void gn_kernel(
    const float* __restrict__ x,
    const float* __restrict__ gx_w, const float* __restrict__ gx_b,
    const float* __restrict__ gn_w, const float* __restrict__ gn_b,
    bf16* __restrict__ hx, bf16* __restrict__ hn) {
    const int blk  = blockIdx.x;          // 0..255
    const int half = blk >> 7;
    const int rem  = blk & 127;
    const int b    = rem >> 5;
    const int g    = rem & 31;
    const int N    = CPG * NL;            // 16384 contiguous floats per group
    const int c0   = g * CPG;

    const float* src = x + ((size_t)b * (2 * NC) + half * NC + c0) * (size_t)NL;
    const float* gw  = half ? gn_w : gx_w;
    const float* gb  = half ? gn_b : gx_b;

    float s = 0.f, sq = 0.f;
    for (int i = threadIdx.x; i < N; i += 256) {
        float v = src[i]; s += v; sq += v * v;
    }
    __shared__ float rs[256], rq[256];
    rs[threadIdx.x] = s; rq[threadIdx.x] = sq;
    __syncthreads();
    for (int off = 128; off > 0; off >>= 1) {
        if (threadIdx.x < off) {
            rs[threadIdx.x] += rs[threadIdx.x + off];
            rq[threadIdx.x] += rq[threadIdx.x + off];
        }
        __syncthreads();
    }
    const float mean = rs[0] / (float)N;
    const float var  = rq[0] / (float)N - mean * mean;
    const float rstd = rsqrtf(var + 1e-6f);

    float gwv[CPG], gbv[CPG];
#pragma unroll
    for (int u = 0; u < CPG; ++u) { gwv[u] = gw[c0 + u]; gbv[u] = gb[c0 + u]; }

    bf16* dstb = (half ? hn : hx) + (size_t)b * NL * NC + c0;   // (L,C)
    for (int l = threadIdx.x; l < NL; l += 256) {
        v4bf pk;
#pragma unroll
        for (int u = 0; u < CPG; ++u)
            pk[u] = (bf16)((src[(size_t)u * NL + l] - mean) * rstd * gwv[u] + gbv[u]);
        *(v4bf*)(dstb + (size_t)l * NC) = pk;   // 8B aligned (c0 multiple of 4)
    }
}

// ---------------------------------------------------------------- q/k/xv/nv projections (LDS-free)
// D[m=l][n=d] = sum_c H[l0+m][c] * W[c][d0+n] + bias[d]
// A[m][k] = H[l0+m][k]        (contiguous 16B row segments, H in (L,C))
// B[k][n] = W[k][d0+n] = Wt[d][k]  (contiguous 32B row segment per lane)
// q,k stored (L,C); xv,nv stored (C,L)
__global__ __launch_bounds__(256) void nin4_kernel(
    const bf16* __restrict__ hx, const bf16* __restrict__ hn,
    const bf16* __restrict__ wbfT,
    const float* __restrict__ bq, const float* __restrict__ bk,
    const float* __restrict__ bxv, const float* __restrict__ bnv,
    bf16* __restrict__ qb, bf16* __restrict__ kbuf,
    bf16* __restrict__ xvb, bf16* __restrict__ nvb) {
    const int tid = threadIdx.x, wv = tid >> 5, lane = tid & 31;
    const int hi = lane >> 4, col = lane & 15;
    const int which = blockIdx.z, b = blockIdx.y;
    const int l0 = blockIdx.x * 128 + 16 * wv;      // wave's 16 l rows

    const bf16*  in   = (which == 0 || which == 2) ? hx : hn;
    const bf16*  wT   = wbfT + (size_t)which * 16384;
    const float* bias = (which == 0) ? bq : (which == 1) ? bk : (which == 2) ? bxv : bnv;
    bf16*        out  = (which == 0) ? qb : (which == 1) ? kbuf : (which == 2) ? xvb : nvb;

    // A fragments: two b128 loads per 32-channel chunk
    const bf16* hrow = in + ((size_t)b * NL + l0 + col) * NC;
    v16bf af[4];
#pragma unroll
    for (int kc = 0; kc < 4; ++kc)
        af[kc] = ld16(hrow + 32 * kc + 16 * hi);

    v8f acc[8] = {};
#pragma unroll
    for (int nt = 0; nt < 8; ++nt) {
        const bf16* wrow = wT + (size_t)(16 * nt + col) * 128;
#pragma unroll
        for (int kc = 0; kc < 4; ++kc) {
            v16bf bfrag = ld16(wrow + 32 * kc + 16 * hi);
            acc[nt] = __builtin_amdgcn_wmma_f32_16x16x32_bf16(
                false, af[kc], false, bfrag, (short)0, acc[nt], false, false);
        }
    }

#pragma unroll
    for (int nt = 0; nt < 8; ++nt) {
        const int d = 16 * nt + col;
        const float bd = bias[d];
#pragma unroll
        for (int r = 0; r < 8; ++r) {
            const int l = l0 + r + 8 * hi;
            const float v = acc[nt][r] + bd;
            if (which < 2) out[((size_t)b * NL + l) * NC + d] = (bf16)v;   // (L,C) coalesced
            else           out[((size_t)(b * NC + d)) * NL + l] = (bf16)v; // (C,L)
        }
    }
}

// ---------------------------------------------------------------- zero loss slot
__global__ void zero_loss_kernel(float* p) { *p = 0.f; }

// ---------------------------------------------------------------- flash attention
// q,k in (L,C); xv,nv in (C,L); outputs (L,C)
#define KP 136    // sK row stride (272B, 16B aligned, bank-skewed)
#define VP 40     // sV/sU/sP row stride (80B, 16B aligned, bank-skewed)

__global__ __launch_bounds__(128) void attn_kernel(
    const bf16* __restrict__ qb, const bf16* __restrict__ kb,
    const bf16* __restrict__ xvb, const bf16* __restrict__ nvb,
    bf16* __restrict__ oxb, bf16* __restrict__ onb) {
    __shared__ bf16 sK[32][KP];        // K tile  [j][c]
    __shared__ bf16 sV[128][VP];       // XV tile [c][j]
    __shared__ bf16 sU[128][VP];       // NV tile [c][j]
    __shared__ bf16 sP[4][16][VP];     // per-wave probability staging [i][j]

    const int tid = threadIdx.x, wv = tid >> 5, lane = tid & 31;
    const int hi = lane >> 4, col = lane & 15;
    const int b = blockIdx.y;
    const int i0 = blockIdx.x * 64 + 16 * wv;

    // Q fragments: two 16B global loads per chunk
    const bf16* qrow = qb + ((size_t)b * NL + i0 + col) * NC;
    v16bf qf[4];
#pragma unroll
    for (int kc = 0; kc < 4; ++kc)
        qf[kc] = ld16(qrow + 32 * kc + 16 * hi);

    v8f accx[8] = {}, accn[8] = {};
    float mrun[8], lrun[8];
#pragma unroll
    for (int r = 0; r < 8; ++r) { mrun[r] = -3.0e38f; lrun[r] = 0.f; }

    for (int j0 = 0; j0 < NL; j0 += 32) {
        __syncthreads();
        {   // K tile: 32 rows x 256B, thread -> (row, 64B segment)
            const int j = tid >> 2, seg = tid & 3;
            const u32x4* srk = (const u32x4*)(kb + ((size_t)b * NL + j0 + j) * NC + seg * 32);
            u32x4* dsk = (u32x4*)(&sK[j][seg * 32]);
#pragma unroll
            for (int u = 0; u < 4; ++u) dsk[u] = srk[u];
            // V/U tiles: 128 rows x 64B, thread -> row
            const u32x4* srv = (const u32x4*)(xvb + ((size_t)(b * NC + tid)) * NL + j0);
            const u32x4* sru = (const u32x4*)(nvb + ((size_t)(b * NC + tid)) * NL + j0);
            u32x4* dsv = (u32x4*)(&sV[tid][0]);
            u32x4* dsu = (u32x4*)(&sU[tid][0]);
#pragma unroll
            for (int u = 0; u < 4; ++u) { dsv[u] = srv[u]; dsu[u] = sru[u]; }
        }
        __syncthreads();

        // S = Q Kt : two 16x16 tiles; B[k][n] = K[j0+n][k] -> contiguous per lane
        v8f s0 = {}, s1 = {};
#pragma unroll
        for (int kc = 0; kc < 4; ++kc) {
            v16bf b0 = ld16(&sK[col][32 * kc + 16 * hi]);
            v16bf b1 = ld16(&sK[16 + col][32 * kc + 16 * hi]);
            s0 = __builtin_amdgcn_wmma_f32_16x16x32_bf16(false, qf[kc], false, b0, (short)0, s0, false, false);
            s1 = __builtin_amdgcn_wmma_f32_16x16x32_bf16(false, qf[kc], false, b1, (short)0, s1, false, false);
        }

        // online softmax per row (row r+8*hi lives across a 16-lane half-group)
        float sc[8];
#pragma unroll
        for (int r = 0; r < 8; ++r) {
            float a0 = s0[r] * ATT_SCALE, a1 = s1[r] * ATT_SCALE;
            float mloc = fmaxf(a0, a1);
#pragma unroll
            for (int off = 8; off > 0; off >>= 1) mloc = fmaxf(mloc, __shfl_xor(mloc, off, 16));
            float mnew = fmaxf(mrun[r], mloc);
            float corr = __expf(mrun[r] - mnew);
            float p0 = __expf(a0 - mnew), p1 = __expf(a1 - mnew);
            float rsum = p0 + p1;
#pragma unroll
            for (int off = 8; off > 0; off >>= 1) rsum += __shfl_xor(rsum, off, 16);
            lrun[r] = lrun[r] * corr + rsum;
            mrun[r] = mnew;
            sc[r] = corr;
            sP[wv][r + 8 * hi][col]      = (bf16)p0;
            sP[wv][r + 8 * hi][16 + col] = (bf16)p1;
        }
#pragma unroll
        for (int t = 0; t < 8; ++t) {
#pragma unroll
            for (int r = 0; r < 8; ++r) { accx[t][r] *= sc[r]; accn[t][r] *= sc[r]; }
        }
        asm volatile("s_wait_dscnt 0x0" ::: "memory");  // D-layout -> A-layout via LDS

        v16bf pf = ld16(&sP[wv][col][16 * hi]);

        // O += P * Vt : B[k=j][n=c] = V[c][j0+k] -> contiguous per lane
#pragma unroll
        for (int t = 0; t < 8; ++t) {
            v16bf bx = ld16(&sV[16 * t + col][16 * hi]);
            v16bf bu = ld16(&sU[16 * t + col][16 * hi]);
            accx[t] = __builtin_amdgcn_wmma_f32_16x16x32_bf16(false, pf, false, bx, (short)0, accx[t], false, false);
            accn[t] = __builtin_amdgcn_wmma_f32_16x16x32_bf16(false, pf, false, bu, (short)0, accn[t], false, false);
        }
    }

    // normalize and store (L,C), coalesced across lanes
#pragma unroll
    for (int r = 0; r < 8; ++r) {
        const float inv = 1.f / lrun[r];
        const size_t rowbase = ((size_t)b * NL + (i0 + r + 8 * hi)) * NC;
#pragma unroll
        for (int t = 0; t < 8; ++t) {
            oxb[rowbase + 16 * t + col] = (bf16)(accx[t][r] * inv);
            onb[rowbase + 16 * t + col] = (bf16)(accn[t][r] * inv);
        }
    }
}

// ---------------------------------------------------------------- output projection + epilogue (LDS-free)
// D[m=l][n=d] = sum_c O[l][c] * Wo[c][d] + bias; diag mask; loss; residual; /sqrt2
__global__ __launch_bounds__(256) void proj_kernel(
    const bf16* __restrict__ oxb, const bf16* __restrict__ onb,
    const bf16* __restrict__ wbfT,
    const float* __restrict__ bxo, const float* __restrict__ bno,
    const float* __restrict__ x,
    float* __restrict__ outp, float* __restrict__ lossp) {
    const int tid = threadIdx.x, wv = tid >> 5, lane = tid & 31;
    const int hi = lane >> 4, col = lane & 15;
    const int l0 = blockIdx.x * 16, which = blockIdx.y, b = blockIdx.z;

    const bf16*  O    = which ? onb : oxb;
    const bf16*  wT   = wbfT + (size_t)(4 + which) * 16384;
    const float* bias = which ? bno : bxo;

    const bf16* orow = O + ((size_t)b * NL + l0 + col) * NC;    // A: O rows (L,C)
    const bf16* wrow = wT + (size_t)(16 * wv + col) * 128;      // B: Wt rows [d][c]

    v8f acc = {};
#pragma unroll
    for (int kc = 0; kc < 4; ++kc) {
        v16bf af    = ld16(orow + 32 * kc + 16 * hi);
        v16bf bfrag = ld16(wrow + 32 * kc + 16 * hi);
        acc = __builtin_amdgcn_wmma_f32_16x16x32_bf16(false, af, false, bfrag, (short)0, acc, false, false);
    }

    const int d = 16 * wv + col;
    const float bd = bias[d];
    float loss = 0.f;
#pragma unroll
    for (int r = 0; r < 8; ++r) {
        const int l = l0 + r + 8 * hi;
        const float val = acc[r] + bd;
        const int hrow = l >> 6, wc = l & 63;
        const bool diag = (hrow == wc);
        const float masked = diag ? val * 1e-12f : val;
        if (!diag) loss += fabsf(val);
        const size_t idx = ((size_t)b * 256 + which * NC + d) * (size_t)NL + l;
        outp[idx] = (x[idx] + masked) * INV_SQRT2;
    }
#pragma unroll
    for (int off = 16; off > 0; off >>= 1) loss += __shfl_xor(loss, off, 32);
    if (lane == 0) atomicAdd(lossp, loss);
}

// ---------------------------------------------------------------- launch
extern "C" void kernel_launch(void* const* d_in, const int* in_sizes, int n_in,
                              void* d_out, int out_size, void* d_ws, size_t ws_size,
                              hipStream_t stream) {
    (void)in_sizes; (void)n_in; (void)out_size; (void)ws_size;
    const float* x     = (const float*)d_in[0];
    const float* gnx_w = (const float*)d_in[1];
    const float* gnx_b = (const float*)d_in[2];
    const float* gnn_w = (const float*)d_in[3];
    const float* gnn_b = (const float*)d_in[4];
    const float* wq    = (const float*)d_in[5];
    const float* bq    = (const float*)d_in[6];
    const float* wk    = (const float*)d_in[7];
    const float* bk    = (const float*)d_in[8];
    const float* wxv   = (const float*)d_in[9];
    const float* bxv   = (const float*)d_in[10];
    const float* wnv   = (const float*)d_in[11];
    const float* bnv   = (const float*)d_in[12];
    const float* wxo   = (const float*)d_in[13];
    const float* bxo   = (const float*)d_in[14];
    const float* wno   = (const float*)d_in[15];
    const float* bno   = (const float*)d_in[16];

    char* ws = (char*)d_ws;
    size_t off = 0;
    auto take = [&](size_t bytes) {
        void* p = ws + off;
        off += (bytes + 255) & ~(size_t)255;
        return p;
    };
    const size_t BCL = (size_t)NB * NC * NL;
    bf16* hx   = (bf16*)take(BCL * 2);   // (L,C)
    bf16* hn   = (bf16*)take(BCL * 2);   // (L,C)
    bf16* qb   = (bf16*)take(BCL * 2);   // (L,C)
    bf16* kbf  = (bf16*)take(BCL * 2);   // (L,C)
    bf16* xvb  = (bf16*)take(BCL * 2);   // (C,L)
    bf16* nvb  = (bf16*)take(BCL * 2);   // (C,L)
    bf16* oxb  = (bf16*)take(BCL * 2);   // (L,C)
    bf16* onb  = (bf16*)take(BCL * 2);   // (L,C)
    bf16* wbfT = (bf16*)take(6 * 16384 * 2);

    float* outp  = (float*)d_out;
    float* lossp = outp + RESULT_ELEMS;

    wconv_kernel<<<dim3(64, 6), 256, 0, stream>>>(wq, wk, wxv, wnv, wxo, wno, wbfT);
    gn_kernel<<<dim3(NB * NG * 2), 256, 0, stream>>>(x, gnx_w, gnx_b, gnn_w, gnn_b, hx, hn);
    nin4_kernel<<<dim3(NL / 128, NB, 4), 256, 0, stream>>>(hx, hn, wbfT, bq, bk, bxv, bnv,
                                                           qb, kbf, xvb, nvb);
    zero_loss_kernel<<<1, 1, 0, stream>>>(lossp);
    attn_kernel<<<dim3(NL / 64, NB), 128, 0, stream>>>(qb, kbf, xvb, nvb, oxb, onb);
    proj_kernel<<<dim3(NL / 16, 2, NB), 256, 0, stream>>>(oxb, onb, wbfT, bxo, bno, x,
                                                          outp, lossp);
}
